// AttnDecoderRNN_73280732004371
// MI455X (gfx1250) — matrix operand order repaired and checked by reading, not compiled, not run on hardware
//
#include <hip/hip_runtime.h>
#include <hip/hip_bf16.h>
#include <math.h>

// Problem constants (match reference).
constexpr int NV = 32000;  // vocab
constexpr int NH = 256;    // hidden
constexpr int NL = 4;      // layers
constexpr int NB = 32;     // batch
constexpr int NS = 64;     // source len
constexpr int NT = 64;     // target len
constexpr int NG = 3 * NH; // 768 gate width

typedef __attribute__((ext_vector_type(16))) _Float16 v16h;
typedef __attribute__((ext_vector_type(8)))  _Float16 v8h;
typedef __attribute__((ext_vector_type(8)))  float    v8f;

// A fragment: lane<16 -> M=lane, halves 0..15 = K kb+0..15;
// lanes>=16 use base +16 halves (K kb+16..31).
__device__ __forceinline__ v16h load_fragA(const _Float16* ap, int kb)
{
    v8h lo = *(const v8h*)(ap + kb);
    v8h hi = *(const v8h*)(ap + kb + 8);
    return __builtin_shufflevector(lo, hi,
            0,1,2,3,4,5,6,7,8,9,10,11,12,13,14,15);
}

// ---------------------------------------------------------------------------
// WMMA GEMM:  C[M,N] = A[M,K] * W[N,K]^T + bias[N]
// A, W are f16 row-major (leading dims in halves). One wave computes a
// 16 x (16*NBLK) strip of C.  K is a compile-time constant so the software-
// pipelined k-loop fully unrolls: the guards constant-fold, loads for block
// kb+32 issue before the WMMAs of kb, and no register-rotation movs remain.
// MODE 0: C[m*ldc + n]   MODE 1: logits scatter (row = t*32+b -> out[b,t,n])
// ---------------------------------------------------------------------------
template <int NBLK, int MODE, int K>
__global__ __launch_bounds__(256)
void k_gemm_f16(const _Float16* __restrict__ Aq, int lda,
                const _Float16* __restrict__ Wq, int ldb,
                const float* __restrict__ bias,
                float* __restrict__ Cc, int ldc,
                int M, int N)
{
    const int lane    = threadIdx.x & 31;
    const int wave    = threadIdx.x >> 5;
    const int groupsN = (N >> 4) / NBLK;
    const int ngroups = (M >> 4) * groupsN;
    const int group   = blockIdx.x * (blockDim.x >> 5) + wave;
    if (group >= ngroups) return;               // wave-uniform: EXEC stays all-1s
    const int m0 = (group / groupsN) << 4;
    const int n0 = (group % groupsN) * (16 * NBLK);

    const _Float16* ap = Aq + (size_t)(m0 + (lane & 15)) * lda + ((lane >> 4) << 4);
    // B fragment: W is [N,K] row-major => column n contiguous in k.
    // lane<16 -> N=lane, K kb..kb+15 ; lane>=16 -> K kb+16..kb+31.
    const _Float16* bp[NBLK];
    #pragma unroll
    for (int j = 0; j < NBLK; ++j)
        bp[j] = Wq + (size_t)(n0 + j * 16 + (lane & 15)) * ldb + ((lane >> 4) << 4);

    v8f acc[NBLK];
    #pragma unroll
    for (int j = 0; j < NBLK; ++j) acc[j] = (v8f){};

    // Software pipeline (fully unrolled; all guards are compile-time).
    v16h a = load_fragA(ap, 0);
    v16h b[NBLK];
    #pragma unroll
    for (int j = 0; j < NBLK; ++j) b[j] = *(const v16h*)(bp[j]);

    #pragma unroll
    for (int kb = 0; kb < K; kb += 32) {
        v16h an;
        v16h bn[NBLK];
        if (kb + 32 < K) {
            an = load_fragA(ap, kb + 32);
            #pragma unroll
            for (int j = 0; j < NBLK; ++j)
                bn[j] = *(const v16h*)(bp[j] + kb + 32);
        }
        #pragma unroll
        for (int j = 0; j < NBLK; ++j)
            acc[j] = __builtin_amdgcn_wmma_f32_16x16x32_f16(
                         false, a, false, b[j], (short)0, acc[j], false, false);
        if (kb + 32 < K) {
            a = an;
            #pragma unroll
            for (int j = 0; j < NBLK; ++j) b[j] = bn[j];
        }
    }

    // C layout (32-bit 16x16): lane<16 -> N=lane, VGPR r = M=r ; lanes>=16 M=r+8.
    const int mbase = m0 + ((lane >> 4) << 3);
    #pragma unroll
    for (int j = 0; j < NBLK; ++j) {
        const int n  = n0 + j * 16 + (lane & 15);
        const float bv = bias ? bias[n] : 0.0f;
        if (MODE == 0) {
            #pragma unroll
            for (int r = 0; r < 8; ++r)
                Cc[(size_t)(mbase + r) * ldc + n] = acc[j][r] + bv;
        } else {
            #pragma unroll
            for (int r = 0; r < 8; ++r) {
                const int row = mbase + r;          // row = t*32 + b
                const int tt = row >> 5, bb = row & 31;
                Cc[((size_t)bb * NT + tt) * NV + n] = acc[j][r] + bv;
            }
        }
    }
}

// ---------------------------------------------------------------------------
// f32 -> f16 conversion
// ---------------------------------------------------------------------------
__global__ void k_f32_to_f16(const float* __restrict__ s, _Float16* __restrict__ d, int n)
{
    int i = blockIdx.x * 256 + threadIdx.x;
    if (i < n) d[i] = (_Float16)s[i];
}

// Embedding gather for all (t,b) at once -> f16 [NT,NB,NH]
__global__ void k_embed(const int* __restrict__ tok, const float* __restrict__ emb,
                        _Float16* __restrict__ out)
{
    const int tb = blockIdx.x;               // t*NB + b
    const int t = tb / NB, b = tb % NB;
    const int id = tok[b * NT + t];
    out[(size_t)tb * NH + threadIdx.x] = (_Float16)emb[(size_t)id * NH + threadIdx.x];
}

// encoder_hidden [L,B,H] f32 -> qh16 f16 laid out as query rows [B, L*H]
__global__ void k_qh_init(const float* __restrict__ eh, _Float16* __restrict__ qh)
{
    const int i = blockIdx.x * 256 + threadIdx.x;      // i < L*B*H
    const int l = i / (NB * NH), r = i % (NB * NH);
    const int b = r / NH, h = r % NH;
    qh[b * (NL * NH) + l * NH + h] = (_Float16)eh[i];
}

// ---------------------------------------------------------------------------
// Attention for one step: scores = Va . tanh(q + UaK), softmax, ctx = w @ keys
// One block per b.  Writes x0 = [emb | ctx] (f16) and attention weights.
// ---------------------------------------------------------------------------
__global__ void k_attn(const float* __restrict__ qf, const float* __restrict__ UaK,
                       const float* __restrict__ keys, const float* __restrict__ Va,
                       const float* __restrict__ bVa,
                       const _Float16* __restrict__ embT,
                       _Float16* __restrict__ x0,
                       float* __restrict__ attnOut, int t)
{
    __shared__ float sc[NS];
    const int b = blockIdx.x;
    const int lane = threadIdx.x & 31;
    const int wave = threadIdx.x >> 5;

    float q8[8], va8[8];
    #pragma unroll
    for (int j = 0; j < 8; ++j) {
        q8[j]  = qf[b * NH + lane * 8 + j];
        va8[j] = Va[lane * 8 + j];
    }
    #pragma unroll
    for (int i = 0; i < 8; ++i) {
        const int s = wave * 8 + i;
        const float* u = UaK + ((size_t)(b * NS + s)) * NH + lane * 8;
        float p = 0.f;
        #pragma unroll
        for (int j = 0; j < 8; ++j) p += va8[j] * tanhf(q8[j] + u[j]);
        #pragma unroll
        for (int off = 16; off > 0; off >>= 1) p += __shfl_down(p, off);
        if (lane == 0) sc[s] = p + bVa[0];
    }
    __syncthreads();
    if (wave == 0) {
        float s0 = sc[lane], s1 = sc[lane + 32];
        float mx = fmaxf(s0, s1);
        #pragma unroll
        for (int off = 16; off > 0; off >>= 1) mx = fmaxf(mx, __shfl_xor(mx, off));
        float e0 = expf(s0 - mx), e1 = expf(s1 - mx);
        float sum = e0 + e1;
        #pragma unroll
        for (int off = 16; off > 0; off >>= 1) sum += __shfl_xor(sum, off);
        const float inv = 1.0f / sum;
        e0 *= inv; e1 *= inv;
        sc[lane] = e0; sc[lane + 32] = e1;
        float* ao = attnOut + (size_t)b * NT * NS + (size_t)t * NS;
        ao[lane] = e0; ao[lane + 32] = e1;
    }
    __syncthreads();
    const int h = threadIdx.x;
    float c = 0.f;
    for (int s = 0; s < NS; ++s)
        c += sc[s] * keys[((size_t)(b * NS + s)) * NH + h];
    x0[b * 2 * NH + NH + h] = (_Float16)c;
    x0[b * 2 * NH + h]      = embT[b * NH + h];
}

// ---------------------------------------------------------------------------
// GRU gates: r=sig(ir+hr), z=sig(iz+hz), n=tanh(in + r*hn), h'=(1-z)n+z h
// Updates f32 hidden in place, writes f16 copies for next GEMMs.
// ---------------------------------------------------------------------------
__global__ void k_gru_gate(const float* __restrict__ Gi, const float* __restrict__ Gh,
                           float* __restrict__ hidden, _Float16* __restrict__ qh,
                           _Float16* __restrict__ xnext, int l)
{
    const int b = blockIdx.x, h = threadIdx.x;
    const float gir = Gi[b * NG + h], giz = Gi[b * NG + NH + h], gin = Gi[b * NG + 2 * NH + h];
    const float ghr = Gh[b * NG + h], ghz = Gh[b * NG + NH + h], ghn = Gh[b * NG + 2 * NH + h];
    const float r = 1.f / (1.f + expf(-(gir + ghr)));
    const float z = 1.f / (1.f + expf(-(giz + ghz)));
    const float n = tanhf(gin + r * ghn);
    float* hp = hidden + (size_t)l * NB * NH + b * NH + h;
    const float hn = (1.f - z) * n + z * (*hp);
    *hp = hn;
    qh[b * (NL * NH) + l * NH + h] = (_Float16)hn;
    xnext[b * NH + h] = (_Float16)hn;
}

// ---------------------------------------------------------------------------
// In-place log_softmax over each V=32000 row. One block per (b,t) row.
// ---------------------------------------------------------------------------
__global__ void k_logsoftmax(float* __restrict__ out)
{
    __shared__ float sred[8];
    __shared__ float sbc;
    float* p = out + (size_t)blockIdx.x * NV;
    const int tid = threadIdx.x, lane = tid & 31, wave = tid >> 5;

    float mx = -3.4e38f;
    for (int i = tid; i < NV; i += 256) mx = fmaxf(mx, p[i]);
    #pragma unroll
    for (int off = 16; off > 0; off >>= 1) mx = fmaxf(mx, __shfl_xor(mx, off));
    if (lane == 0) sred[wave] = mx;
    __syncthreads();
    if (tid == 0) { float m = sred[0]; for (int i = 1; i < 8; ++i) m = fmaxf(m, sred[i]); sbc = m; }
    __syncthreads();
    mx = sbc;

    float sum = 0.f;
    for (int i = tid; i < NV; i += 256) sum += expf(p[i] - mx);
    #pragma unroll
    for (int off = 16; off > 0; off >>= 1) sum += __shfl_xor(sum, off);
    __syncthreads();
    if (lane == 0) sred[wave] = sum;
    __syncthreads();
    if (tid == 0) { float s = 0.f; for (int i = 0; i < 8; ++i) s += sred[i]; sbc = logf(s); }
    __syncthreads();
    const float ls = sbc;
    for (int i = tid; i < NV; i += 256) p[i] = p[i] - mx - ls;
}

// ---------------------------------------------------------------------------
template <int K>
static inline void gemmK(hipStream_t st, const _Float16* A, int lda,
                         const _Float16* W, int ldb, const float* bias,
                         float* C, int ldc, int M, int N)
{
    const int ntiles = (M >> 4) * (N >> 4);
    const int blocks = (ntiles + 7) / 8;
    k_gemm_f16<1, 0, K><<<blocks, 256, 0, st>>>(A, lda, W, ldb, bias, C, ldc, M, N);
}

extern "C" void kernel_launch(void* const* d_in, const int* in_sizes, int n_in,
                              void* d_out, int out_size, void* d_ws, size_t ws_size,
                              hipStream_t stream)
{
    const float* keys   = (const float*)d_in[0];   // [B,S,H]
    const float* encH   = (const float*)d_in[1];   // [L,B,H]
    const int*   decIn  = (const int*)  d_in[2];   // [B,T]
    const float* embF   = (const float*)d_in[3];   // [V,H]
    const float* WaF    = (const float*)d_in[4];   // [H, L*H]
    const float* bWaF   = (const float*)d_in[5];
    const float* UaF    = (const float*)d_in[6];   // [H,H]
    const float* bUaF   = (const float*)d_in[7];
    const float* VaF    = (const float*)d_in[8];   // [1,H]
    const float* bVaF   = (const float*)d_in[9];
    const float* Wih0F  = (const float*)d_in[10];  // [768,512]
    const float* WihRF  = (const float*)d_in[11];  // [3,768,256]
    const float* WhhF   = (const float*)d_in[12];  // [4,768,256]
    const float* bihF   = (const float*)d_in[13];  // [4,768]
    const float* bhhF   = (const float*)d_in[14];  // [4,768]
    const float* WoutF  = (const float*)d_in[15];  // [V,H]
    const float* boutF  = (const float*)d_in[16];  // [V]
    float* out = (float*)d_out;

    char* ws = (char*)d_ws;
    size_t off = 0;
    auto alloc = [&](size_t bytes) -> void* {
        size_t o = (off + 255) & ~(size_t)255;
        off = o + bytes;
        return (void*)(ws + o);
    };
    _Float16* Ua16    = (_Float16*)alloc((size_t)NH * NH * 2);
    _Float16* Wa16    = (_Float16*)alloc((size_t)NH * NL * NH * 2);
    _Float16* Wih0_16 = (_Float16*)alloc((size_t)NG * 2 * NH * 2);
    _Float16* WihR16  = (_Float16*)alloc((size_t)3 * NG * NH * 2);
    _Float16* Whh16   = (_Float16*)alloc((size_t)NL * NG * NH * 2);
    _Float16* Wout16  = (_Float16*)alloc((size_t)NV * NH * 2);
    _Float16* keys16  = (_Float16*)alloc((size_t)NB * NS * NH * 2);
    _Float16* emb16   = (_Float16*)alloc((size_t)NT * NB * NH * 2);
    float*    UaKb    = (float*)   alloc((size_t)NB * NS * NH * 4);
    float*    hid     = (float*)   alloc((size_t)NL * NB * NH * 4);
    _Float16* qh16    = (_Float16*)alloc((size_t)NB * NL * NH * 2);
    float*    qf      = (float*)   alloc((size_t)NB * NH * 4);
    _Float16* x0      = (_Float16*)alloc((size_t)NB * 2 * NH * 2);
    _Float16* xl      = (_Float16*)alloc((size_t)NB * NH * 2);
    float*    Gi      = (float*)   alloc((size_t)NB * NG * 4);
    float*    Gh      = (float*)   alloc((size_t)NB * NG * 4);
    _Float16* xall    = (_Float16*)alloc((size_t)NT * NB * NH * 2);
    (void)ws_size; (void)n_in; (void)in_sizes; (void)out_size;

    auto cvt = [&](const float* s, _Float16* d, size_t n) {
        k_f32_to_f16<<<(unsigned)((n + 255) / 256), 256, 0, stream>>>(s, d, (int)n);
    };
    // One-time weight / input conversions to f16.
    cvt(UaF,   Ua16,    (size_t)NH * NH);
    cvt(WaF,   Wa16,    (size_t)NH * NL * NH);
    cvt(Wih0F, Wih0_16, (size_t)NG * 2 * NH);
    cvt(WihRF, WihR16,  (size_t)3 * NG * NH);
    cvt(WhhF,  Whh16,   (size_t)NL * NG * NH);
    cvt(WoutF, Wout16,  (size_t)NV * NH);
    cvt(keys,  keys16,  (size_t)NB * NS * NH);
    k_embed<<<NT * NB, NH, 0, stream>>>(decIn, embF, emb16);
    k_qh_init<<<(NL * NB * NH) / 256, 256, 0, stream>>>(encH, qh16);
    hipMemcpyAsync(hid, encH, (size_t)NL * NB * NH * 4, hipMemcpyDeviceToDevice, stream);

    // UaK = keys @ Ua^T + bUa : [2048,256] x [256,256]
    gemmK<256>(stream, keys16, NH, Ua16, NH, bUaF, UaKb, NH, NB * NS, NH);

    float* attnBase = out + (size_t)NB * NT * NV + (size_t)NL * NB * NH;

    for (int t = 0; t < NT; ++t) {
        // q = query @ Wa^T + bWa : [32,256] with K = L*H = 1024
        gemmK<1024>(stream, qh16, NL * NH, Wa16, NL * NH, bWaF, qf, NH, NB, NH);
        k_attn<<<NB, NH, 0, stream>>>(qf, UaKb, keys, VaF, bVaF,
                                      emb16 + (size_t)t * NB * NH, x0, attnBase, t);
        for (int l = 0; l < NL; ++l) {
            if (l == 0) {
                gemmK<512>(stream, x0, 2 * NH, Wih0_16, 2 * NH, bihF, Gi, NG, NB, NG);
            } else {
                gemmK<256>(stream, xl, NH, WihR16 + (size_t)(l - 1) * NG * NH, NH,
                           bihF + l * NG, Gi, NG, NB, NG);
            }
            gemmK<256>(stream, qh16 + l * NH, NL * NH, Whh16 + (size_t)l * NG * NH, NH,
                       bhhF + l * NG, Gh, NG, NB, NG);
            _Float16* xn = (l == NL - 1) ? (xall + (size_t)t * NB * NH) : xl;
            k_gru_gate<<<NB, NH, 0, stream>>>(Gi, Gh, hid, qh16, xn, l);
        }
    }

    // logits = X @ W_out^T + b_out : [2048,256] x [256,32000]
    // 4-way N register blocking: 128 M-tiles x 500 groups of 4 N-tiles.
    {
        const int ngroups = (NB * NT >> 4) * ((NV >> 4) / 4);   // 64000
        const int blocks  = (ngroups + 7) / 8;                  // 8000
        k_gemm_f16<4, 1, 256><<<blocks, 256, 0, stream>>>(
            xall, NH, Wout16, NH, boutF, out, NV, NB * NT, NV);
    }
    k_logsoftmax<<<NB * NT, 256, 0, stream>>>(out);
    // hidden_final
    hipMemcpyAsync(out + (size_t)NB * NT * NV, hid,
                   (size_t)NL * NB * NH * 4, hipMemcpyDeviceToDevice, stream);
}
